// MoELayer_37623913513260
// MI455X (gfx1250) — compile-verified, matrix operand order checked by reading, not run on hardware
//
#include <hip/hip_runtime.h>
#include <math.h>

#define D_MODEL 1024
#define D_FF    4096
#define NEXP    8
#define NTOK    8192           // B*T = 4*2048
#define TOPK    2
#define BM      128
#define BN      128
#define BK      32
#define PADCAP  (NTOK*TOPK + NEXP*BM)   // 17408 padded gathered rows
#define MAXTILES (PADCAP/BM)            // 136
#define LDSTR   40                       // ushort stride (80B, multiple of 16B)

// ---- workspace layout (bytes) ----
#define OFF_CNT     0                               // int[8]
#define OFF_CURSOR  64                              // int[8]
#define OFF_PSUM    128                             // float[8]
#define OFF_POFF    192                             // int[9]
#define OFF_ZERO    256                             // 128B zero page for padded gather rows
#define OFF_IDX     1024                            // int[NTOK*2]
#define OFF_GATE    (OFF_IDX   + NTOK*TOPK*4)       // float[NTOK*2]
#define OFF_TOK     (OFF_GATE  + NTOK*TOPK*4)       // int[PADCAP]
#define OFF_RGATE   (OFF_TOK   + PADCAP*4)          // float[PADCAP]
#define OFF_XB      (1u<<20)                        // ushort[NTOK*D_MODEL]  (16MB)
#define OFF_HB      (OFF_XB + NTOK*D_MODEL*2u)      // ushort[PADCAP*D_FF]   (~143MB)

typedef __attribute__((ext_vector_type(16))) __bf16 v16bf;
typedef __attribute__((ext_vector_type(8)))  float  v8f;

union BFrag {
    v16bf v;
    uint4 q[2];
};

__device__ __forceinline__ unsigned short f2bf(float f) {
    unsigned int u = __float_as_uint(f);
    unsigned int r = u + 0x7FFFu + ((u >> 16) & 1u);   // round to nearest even
    return (unsigned short)(r >> 16);
}

__device__ __forceinline__ unsigned pack2(float a, float b) {
    return (unsigned)f2bf(a) | ((unsigned)f2bf(b) << 16);
}

__device__ __forceinline__ float gelu_exact(float v) {
    return 0.5f * v * (1.0f + erff(v * 0.70710678118654752f));
}

// CDNA5 async copy: 16B global -> LDS, tracked by ASYNCcnt (bypasses VGPRs).
// LDS offset = low 32 bits of the generic shared-memory address (ISA 10.2).
__device__ __forceinline__ void async_copy_b128(void* lds_ptr, const void* gptr) {
    unsigned loff = (unsigned)(size_t)lds_ptr;
    asm volatile("global_load_async_to_lds_b128 %0, %1, off"
                 :: "v"(loff), "v"(gptr)
                 : "memory");
}
__device__ __forceinline__ void wait_async0() {
    asm volatile("s_wait_asynccnt 0x0" ::: "memory");
}

// ---------------- init: zero output & counters, poison token list ----------------
__global__ void init_kernel(float* __restrict__ out, int* __restrict__ cnt,
                            int* __restrict__ cursor, float* __restrict__ psum,
                            int* __restrict__ tokl, int* __restrict__ zpage) {
    int i = blockIdx.x * blockDim.x + threadIdx.x;
    if (i < NTOK * D_MODEL) out[i] = 0.0f;
    if (i < PADCAP) tokl[i] = -1;
    if (i < 32) zpage[i] = 0;
    if (i < NEXP) { cnt[i] = 0; cursor[i] = 0; psum[i] = 0.0f; }
}

// ---------------- x -> bf16 once (8 elements per thread) ----------------
__global__ void xcvt_kernel(const float* __restrict__ x, unsigned short* __restrict__ xb) {
    int i = blockIdx.x * blockDim.x + threadIdx.x;   // one per 8 elements
    size_t base = (size_t)i * 8;
    float4 a = *(const float4*)(x + base);
    float4 b = *(const float4*)(x + base + 4);
    uint4 o;
    o.x = pack2(a.x, a.y);
    o.y = pack2(a.z, a.w);
    o.z = pack2(b.x, b.y);
    o.w = pack2(b.z, b.w);
    *(uint4*)(xb + base) = o;
}

// ---------------- router: one wave32 per token ----------------
__global__ __launch_bounds__(256)
void router_kernel(const float* __restrict__ x, const float* __restrict__ Wr,
                   int* __restrict__ cnt, float* __restrict__ psum,
                   int* __restrict__ idx, float* __restrict__ gate) {
    int wave = threadIdx.x >> 5;
    int lane = threadIdx.x & 31;
    int t = blockIdx.x * 8 + wave;
    if (t >= NTOK) return;

    float acc[NEXP];
#pragma unroll
    for (int e = 0; e < NEXP; e++) acc[e] = 0.0f;
    const float* xr = x + (size_t)t * D_MODEL;
    for (int d = lane; d < D_MODEL; d += 32) {
        float xv = xr[d];
#pragma unroll
        for (int e = 0; e < NEXP; e++) acc[e] += xv * Wr[d * NEXP + e];
    }
#pragma unroll
    for (int e = 0; e < NEXP; e++)
        for (int o = 16; o > 0; o >>= 1) acc[e] += __shfl_xor(acc[e], o, 32);

    if (lane == 0) {
        float m = acc[0];
#pragma unroll
        for (int e = 1; e < NEXP; e++) m = fmaxf(m, acc[e]);
        float p[NEXP], s = 0.0f;
#pragma unroll
        for (int e = 0; e < NEXP; e++) { p[e] = __expf(acc[e] - m); s += p[e]; }
        float inv = 1.0f / s;
#pragma unroll
        for (int e = 0; e < NEXP; e++) p[e] *= inv;
        // top-2
        int e1 = 0;
#pragma unroll
        for (int e = 1; e < NEXP; e++) if (p[e] > p[e1]) e1 = e;
        int e2 = (e1 == 0) ? 1 : 0;
#pragma unroll
        for (int e = 0; e < NEXP; e++) if (e != e1 && p[e] > p[e2]) e2 = e;
        float g1 = p[e1], g2 = p[e2], gs = g1 + g2;
        g1 /= gs; g2 /= gs;
        idx[t * 2] = e1; idx[t * 2 + 1] = e2;
        gate[t * 2] = g1; gate[t * 2 + 1] = g2;
        atomicAdd(&cnt[e1], 1);
        atomicAdd(&cnt[e2], 1);
#pragma unroll
        for (int e = 0; e < NEXP; e++) atomicAdd(&psum[e], p[e]);
    }
}

// ---------------- finalize: padded segment offsets + aux loss ----------------
__global__ void finalize_kernel(const int* __restrict__ cnt, const float* __restrict__ psum,
                                int* __restrict__ poff, float* __restrict__ aux_out) {
    if (threadIdx.x == 0 && blockIdx.x == 0) {
        int off = 0;
        float aux = 0.0f;
        for (int e = 0; e < NEXP; e++) {
            poff[e] = off;
            int c = cnt[e];
            off += ((c + BM - 1) / BM) * BM;
            aux += ((float)c / (float)NTOK) * (psum[e] / (float)NTOK);
        }
        poff[NEXP] = off;
        aux_out[0] = 0.01f * (float)NEXP * aux;
    }
}

// ---------------- scatter tokens into padded expert segments ----------------
__global__ void scatter_kernel(const int* __restrict__ idx, const float* __restrict__ gate,
                               const int* __restrict__ poff, int* __restrict__ cursor,
                               int* __restrict__ tokl, float* __restrict__ rgate) {
    int t = blockIdx.x * blockDim.x + threadIdx.x;
    if (t >= NTOK) return;
#pragma unroll
    for (int k = 0; k < TOPK; k++) {
        int e = idx[t * 2 + k];
        int slot = atomicAdd(&cursor[e], 1);
        int r = poff[e] + slot;
        tokl[r] = t;
        rgate[r] = gate[t * 2 + k];
    }
}

// ---------------- GEMM1: h = gelu(xb_gathered @ W1[e] + b1[e]) -> bf16 ----------------
__global__ __launch_bounds__(256)
void gemm1_kernel(const unsigned short* __restrict__ xb, const float* __restrict__ W1,
                  const float* __restrict__ b1, const int* __restrict__ poff,
                  const int* __restrict__ tokl, const unsigned short* __restrict__ zpage,
                  unsigned short* __restrict__ hb) {
    __shared__ __attribute__((aligned(16))) unsigned short lA[BM * LDSTR];
    __shared__ __attribute__((aligned(16))) unsigned short lB[BN * LDSTR];
    __shared__ int lTok[BM];

    int row_base = blockIdx.x * BM;
    if (row_base >= poff[NEXP]) return;
    int e = 0;
    while (e < NEXP - 1 && row_base >= poff[e + 1]) e++;
    int n_base = blockIdx.y * BN;
    const float* W = W1 + (size_t)e * D_MODEL * D_FF;

    int tid = threadIdx.x;
    if (tid < BM) lTok[tid] = tokl[row_base + tid];
    __syncthreads();

    int wave = tid >> 5, lane = tid & 31;
    int wm = wave & 3, wn = wave >> 2;
    int lhalf = (lane >> 4) & 1;
    int l15 = lane & 15;
    int kb = lhalf * 8;

    v8f acc[2][4];
#pragma unroll
    for (int mt = 0; mt < 2; mt++)
#pragma unroll
        for (int nt = 0; nt < 4; nt++) acc[mt][nt] = (v8f){0,0,0,0,0,0,0,0};

    for (int ks = 0; ks < D_MODEL / BK; ks++) {
        int k0 = ks * BK;
        // A: gathered bf16 rows -> LDS via async copy (ASYNCcnt path)
        for (int q = tid; q < BM * BK / 8; q += 256) {   // 512 x 16B
            int r = q >> 2;
            int j = (q & 3) * 8;
            int tk = lTok[r];
            const unsigned short* src =
                (tk >= 0) ? (xb + (size_t)tk * D_MODEL + k0 + j) : (zpage + j);
            async_copy_b128(&lA[r * LDSTR + j], src);
        }
        // B: W1[k][n] -> lB[n][k] (transpose + f32->bf16)
        for (int q = tid; q < BK * BN / 4; q += 256) {
            int kk = q >> 5;
            int n0 = (q & 31) * 4;
            float4 v = *(const float4*)(W + (size_t)(k0 + kk) * D_FF + n_base + n0);
            lB[(n0 + 0) * LDSTR + kk] = f2bf(v.x);
            lB[(n0 + 1) * LDSTR + kk] = f2bf(v.y);
            lB[(n0 + 2) * LDSTR + kk] = f2bf(v.z);
            lB[(n0 + 3) * LDSTR + kk] = f2bf(v.w);
        }
        if (ks + 1 < D_MODEL / BK)
            __builtin_prefetch(W + (size_t)(k0 + BK) * D_FF + n_base + (tid & 31) * 4, 0, 1);
        wait_async0();
        __syncthreads();

        BFrag a[2];
#pragma unroll
        for (int mt = 0; mt < 2; mt++) {
            int m = wm * 32 + mt * 16 + l15;
            a[mt].q[0] = *(const uint4*)&lA[m * LDSTR + kb];
            a[mt].q[1] = *(const uint4*)&lA[m * LDSTR + kb + 16];
        }
#pragma unroll
        for (int nt = 0; nt < 4; nt++) {
            BFrag b;
            int n = wn * 64 + nt * 16 + l15;
            b.q[0] = *(const uint4*)&lB[n * LDSTR + kb];
            b.q[1] = *(const uint4*)&lB[n * LDSTR + kb + 16];
#pragma unroll
            for (int mt = 0; mt < 2; mt++)
                acc[mt][nt] = __builtin_amdgcn_wmma_f32_16x16x32_bf16(
                    false, a[mt].v, false, b.v, (short)0, acc[mt][nt], false, false);
        }
        __syncthreads();
    }

    const float* be = b1 + (size_t)e * D_FF;
#pragma unroll
    for (int mt = 0; mt < 2; mt++)
#pragma unroll
        for (int nt = 0; nt < 4; nt++) {
            int c = n_base + wn * 64 + nt * 16 + l15;
            float bias = be[c];
#pragma unroll
            for (int v = 0; v < 8; v++) {
                int r = row_base + wm * 32 + mt * 16 + v + lhalf * 8;
                float h = acc[mt][nt][v] + bias;
                hb[(size_t)r * D_FF + c] = f2bf(gelu_exact(h));
            }
        }
}

// ---------------- GEMM2: out[token] += gate * (h @ W2[e] + b2[e]) ----------------
__global__ __launch_bounds__(256)
void gemm2_kernel(const unsigned short* __restrict__ hb, const float* __restrict__ W2,
                  const float* __restrict__ b2, const int* __restrict__ poff,
                  const int* __restrict__ tokl, const float* __restrict__ rgate,
                  float* __restrict__ out) {
    __shared__ __attribute__((aligned(16))) unsigned short lA[BM * LDSTR];
    __shared__ __attribute__((aligned(16))) unsigned short lB[BN * LDSTR];
    __shared__ int   lTok[BM];
    __shared__ float lGate[BM];

    int row_base = blockIdx.x * BM;
    if (row_base >= poff[NEXP]) return;
    int e = 0;
    while (e < NEXP - 1 && row_base >= poff[e + 1]) e++;
    int n_base = blockIdx.y * BN;
    const float* W = W2 + (size_t)e * D_FF * D_MODEL;

    int tid = threadIdx.x;
    if (tid < BM) { lTok[tid] = tokl[row_base + tid]; lGate[tid] = rgate[row_base + tid]; }
    __syncthreads();

    int wave = tid >> 5, lane = tid & 31;
    int wm = wave & 3, wn = wave >> 2;
    int lhalf = (lane >> 4) & 1;
    int l15 = lane & 15;
    int kb = lhalf * 8;

    v8f acc[2][4];
#pragma unroll
    for (int mt = 0; mt < 2; mt++)
#pragma unroll
        for (int nt = 0; nt < 4; nt++) acc[mt][nt] = (v8f){0,0,0,0,0,0,0,0};

    for (int ks = 0; ks < D_FF / BK; ks++) {
        int k0 = ks * BK;
        // A: bf16 h rows -> LDS via async copy (all padded rows are valid data)
        for (int q = tid; q < BM * BK / 8; q += 256) {
            int r = q >> 2;
            int j = (q & 3) * 8;
            async_copy_b128(&lA[r * LDSTR + j],
                            hb + (size_t)(row_base + r) * D_FF + k0 + j);
        }
        // B: W2[k][n] -> lB[n][k]
        for (int q = tid; q < BK * BN / 4; q += 256) {
            int kk = q >> 5;
            int n0 = (q & 31) * 4;
            float4 v = *(const float4*)(W + (size_t)(k0 + kk) * D_MODEL + n_base + n0);
            lB[(n0 + 0) * LDSTR + kk] = f2bf(v.x);
            lB[(n0 + 1) * LDSTR + kk] = f2bf(v.y);
            lB[(n0 + 2) * LDSTR + kk] = f2bf(v.z);
            lB[(n0 + 3) * LDSTR + kk] = f2bf(v.w);
        }
        if (ks + 1 < D_FF / BK)
            __builtin_prefetch(W + (size_t)(k0 + BK) * D_MODEL + n_base + (tid & 31) * 4, 0, 1);
        wait_async0();
        __syncthreads();

        BFrag a[2];
#pragma unroll
        for (int mt = 0; mt < 2; mt++) {
            int m = wm * 32 + mt * 16 + l15;
            a[mt].q[0] = *(const uint4*)&lA[m * LDSTR + kb];
            a[mt].q[1] = *(const uint4*)&lA[m * LDSTR + kb + 16];
        }
#pragma unroll
        for (int nt = 0; nt < 4; nt++) {
            BFrag b;
            int n = wn * 64 + nt * 16 + l15;
            b.q[0] = *(const uint4*)&lB[n * LDSTR + kb];
            b.q[1] = *(const uint4*)&lB[n * LDSTR + kb + 16];
#pragma unroll
            for (int mt = 0; mt < 2; mt++)
                acc[mt][nt] = __builtin_amdgcn_wmma_f32_16x16x32_bf16(
                    false, a[mt].v, false, b.v, (short)0, acc[mt][nt], false, false);
        }
        __syncthreads();
    }

    const float* be = b2 + (size_t)e * D_MODEL;
#pragma unroll
    for (int mt = 0; mt < 2; mt++)
#pragma unroll
        for (int nt = 0; nt < 4; nt++) {
            int c = n_base + wn * 64 + nt * 16 + l15;
            float bias = be[c];
#pragma unroll
            for (int v = 0; v < 8; v++) {
                int rl = wm * 32 + mt * 16 + v + lhalf * 8;
                int tk = lTok[rl];
                if (tk >= 0) {
                    float y = acc[mt][nt][v] + bias;
                    atomicAdd(&out[(size_t)tk * D_MODEL + c], lGate[rl] * y);
                }
            }
        }
}

extern "C" void kernel_launch(void* const* d_in, const int* in_sizes, int n_in,
                              void* d_out, int out_size, void* d_ws, size_t ws_size,
                              hipStream_t stream) {
    const float* x  = (const float*)d_in[0];
    const float* Wr = (const float*)d_in[1];
    const float* W1 = (const float*)d_in[2];
    const float* b1 = (const float*)d_in[3];
    const float* W2 = (const float*)d_in[4];
    const float* b2 = (const float*)d_in[5];
    float* out = (float*)d_out;

    char* ws = (char*)d_ws;
    int*   cnt    = (int*)  (ws + OFF_CNT);
    int*   cursor = (int*)  (ws + OFF_CURSOR);
    float* psum   = (float*)(ws + OFF_PSUM);
    int*   poff   = (int*)  (ws + OFF_POFF);
    int*   zpage  = (int*)  (ws + OFF_ZERO);
    int*   idx    = (int*)  (ws + OFF_IDX);
    float* gate   = (float*)(ws + OFF_GATE);
    int*   tokl   = (int*)  (ws + OFF_TOK);
    float* rgate  = (float*)(ws + OFF_RGATE);
    unsigned short* xb = (unsigned short*)(ws + OFF_XB);
    unsigned short* hb = (unsigned short*)(ws + OFF_HB);

    init_kernel<<<(NTOK * D_MODEL) / 256, 256, 0, stream>>>(out, cnt, cursor, psum, tokl, zpage);
    xcvt_kernel<<<(NTOK * D_MODEL / 8) / 256, 256, 0, stream>>>(x, xb);
    router_kernel<<<NTOK / 8, 256, 0, stream>>>(x, Wr, cnt, psum, idx, gate);
    finalize_kernel<<<1, 32, 0, stream>>>(cnt, psum, poff, out + (size_t)NTOK * D_MODEL);
    scatter_kernel<<<NTOK / 256, 256, 0, stream>>>(idx, gate, poff, cursor, tokl, rgate);
    gemm1_kernel<<<dim3(MAXTILES, D_FF / BN), 256, 0, stream>>>(xb, W1, b1, poff, tokl,
                                                                (const unsigned short*)zpage, hb);
    gemm2_kernel<<<dim3(MAXTILES, D_MODEL / BN), 256, 0, stream>>>(hb, W2, b2, poff, tokl, rgate, out);
}
